// vanilla_transformer_block_40673340293378
// MI455X (gfx1250) — matrix operand order repaired and checked by reading, not compiled, not run on hardware
//
#include <hip/hip_runtime.h>
#include <hip/hip_bf16.h>

typedef __bf16 bf16;
typedef __attribute__((ext_vector_type(16))) __bf16 v16bf;
typedef __attribute__((ext_vector_type(8)))  float  v8f;
typedef int v4i __attribute__((__vector_size__(16)));

union ABfrag { uint4 q[2]; v16bf v; };

#define AS1 __attribute__((address_space(1)))
#define AS3 __attribute__((address_space(3)))

#if defined(__has_builtin)
#if __has_builtin(__builtin_amdgcn_global_load_async_to_lds_b128) && \
    __has_builtin(__builtin_amdgcn_s_wait_asynccnt)
#define USE_ASYNC_LDS 1
#endif
#endif

__device__ __forceinline__ bf16 f2bf(float f) {
  unsigned u = __builtin_bit_cast(unsigned, f);
  unsigned r = u + 0x7FFFu + ((u >> 16) & 1u);   // round-to-nearest-even
  return __builtin_bit_cast(bf16, (unsigned short)(r >> 16));
}
__device__ __forceinline__ float bf2f(bf16 x) {
  unsigned short h = __builtin_bit_cast(unsigned short, x);
  return __builtin_bit_cast(float, ((unsigned)h) << 16);
}

#if USE_ASYNC_LDS
__device__ __forceinline__ void async_cp16(const bf16* g, bf16* l) {
  __builtin_amdgcn_global_load_async_to_lds_b128(
      (AS1 v4i*)g, (AS3 v4i*)l, 0, 0);
}
#endif

__global__ __launch_bounds__(256) void cvt_f32_bf16(const float* __restrict__ s,
                                                    bf16* __restrict__ d, long n) {
  long i = (long)blockIdx.x * 1024 + threadIdx.x;
#pragma unroll
  for (int k = 0; k < 4; ++k) {
    long j = i + (long)k * 256;
    if (j < n) d[j] = f2bf(s[j]);
  }
}

// transpose + convert: src (K x N) f32 row-major (row stride ld, col offset colOfs)
// -> dst (N x K) bf16 row-major.  grid = (N/32, K/32), block = 256 (32x8).
__global__ __launch_bounds__(256) void transpose_cvt(const float* __restrict__ src,
                                                     bf16* __restrict__ dst,
                                                     long ld, long colOfs, long K) {
  __shared__ float t[32][33];
  const int tx = threadIdx.x & 31, ty = threadIdx.x >> 5;
  const long n0 = (long)blockIdx.x * 32, k0 = (long)blockIdx.y * 32;
#pragma unroll
  for (int j = 0; j < 4; ++j)
    t[ty + j * 8][tx] = src[(k0 + ty + j * 8) * ld + colOfs + n0 + tx];
  __syncthreads();
#pragma unroll
  for (int j = 0; j < 4; ++j)
    dst[(n0 + ty + j * 8) * K + k0 + tx] = f2bf(t[tx][ty + j * 8]);
}

// ---------------- batched WMMA GEMM, B pre-transposed (N x K) ----------------
// C = epi(alpha * A @ Bt^T + bias).  A: bf16 MxK row-major (lda).
// Bt: bf16 NxK row-major (ldb).  Tile 128x128, 8 waves (4M x 2N), wave = 2x4
// WMMA 16x16x32 tiles.  LDS double-buffered; staging via async copy if available.
#define GBM 128
#define GBN 128
#define GBK 32

template <int MODE>  // 0:f32 plain 1:bf16 plain 2:qk head-scatter 3:o head-gather 4:v scatterT
__global__ __launch_bounds__(256) void gemm_nt_wmma(
    const bf16* __restrict__ A, long lda, long strideA,
    const bf16* __restrict__ Bt, long ldb, long strideB,
    void* __restrict__ Cp, long strideC, const float* __restrict__ bias,
    float alpha, int N, int K, int batchOfs, int doRelu) {
  __shared__ bf16 sA[2][GBM * GBK];
  __shared__ bf16 sB[2][GBN * GBK];
  const int tid  = threadIdx.x;
  const int lane = tid & 31, wave = tid >> 5;
  const int waveM = wave & 3, waveN = wave >> 2;
  const long rowBase = (long)blockIdx.y * GBM;
  const long colBase = (long)blockIdx.x * GBN;
  const bf16* Ab = A  + (long)blockIdx.z * strideA;
  const bf16* Bb = Bt + (long)blockIdx.z * strideB;

  // per-thread staging chunks: rows r0 and r0+64, 16B along K
  const int r0  = tid >> 2;
  const int kc0 = (tid & 3) * 8;

#if !USE_ASYNC_LDS
  uint4 rg[4];
#endif

  v8f acc[2][4] = {};
  const int rsel = lane & 15, khalf = lane >> 4;
  int buf = 0;

#if USE_ASYNC_LDS
  async_cp16(Ab + (rowBase + r0) * lda + kc0,        sA[0] + r0 * GBK + kc0);
  async_cp16(Ab + (rowBase + r0 + 64) * lda + kc0,   sA[0] + (r0 + 64) * GBK + kc0);
  async_cp16(Bb + (colBase + r0) * ldb + kc0,        sB[0] + r0 * GBK + kc0);
  async_cp16(Bb + (colBase + r0 + 64) * ldb + kc0,   sB[0] + (r0 + 64) * GBK + kc0);
#else
  rg[0] = *(const uint4*)(Ab + (rowBase + r0) * lda + kc0);
  rg[1] = *(const uint4*)(Ab + (rowBase + r0 + 64) * lda + kc0);
  rg[2] = *(const uint4*)(Bb + (colBase + r0) * ldb + kc0);
  rg[3] = *(const uint4*)(Bb + (colBase + r0 + 64) * ldb + kc0);
  *(uint4*)(sA[0] + r0 * GBK + kc0)        = rg[0];
  *(uint4*)(sA[0] + (r0 + 64) * GBK + kc0) = rg[1];
  *(uint4*)(sB[0] + r0 * GBK + kc0)        = rg[2];
  *(uint4*)(sB[0] + (r0 + 64) * GBK + kc0) = rg[3];
#endif

  for (int k0 = 0; k0 < K; k0 += GBK) {
    const int nk = k0 + GBK;
#if USE_ASYNC_LDS
    if (nk < K) {  // issue next tile into the other buffer, retire current tile
      async_cp16(Ab + (rowBase + r0) * lda + nk + kc0,      sA[buf ^ 1] + r0 * GBK + kc0);
      async_cp16(Ab + (rowBase + r0 + 64) * lda + nk + kc0, sA[buf ^ 1] + (r0 + 64) * GBK + kc0);
      async_cp16(Bb + (colBase + r0) * ldb + nk + kc0,      sB[buf ^ 1] + r0 * GBK + kc0);
      async_cp16(Bb + (colBase + r0 + 64) * ldb + nk + kc0, sB[buf ^ 1] + (r0 + 64) * GBK + kc0);
      __builtin_amdgcn_s_wait_asynccnt(4);
    } else {
      __builtin_amdgcn_s_wait_asynccnt(0);
    }
    __syncthreads();
#else
    __syncthreads();                 // make commits to sX[buf] visible
    if (nk < K) {                    // prefetch next tile into registers
      rg[0] = *(const uint4*)(Ab + (rowBase + r0) * lda + nk + kc0);
      rg[1] = *(const uint4*)(Ab + (rowBase + r0 + 64) * lda + nk + kc0);
      rg[2] = *(const uint4*)(Bb + (colBase + r0) * ldb + nk + kc0);
      rg[3] = *(const uint4*)(Bb + (colBase + r0 + 64) * ldb + nk + kc0);
    }
#endif

    // ---- fragments per CDNA5 16-bit WMMA layouts ----
    ABfrag aF[2], bF[4];
#pragma unroll
    for (int ms = 0; ms < 2; ++ms) {
      const bf16* base = sA[buf] + (waveM * 32 + ms * 16 + rsel) * GBK + khalf * 8;
      aF[ms].q[0] = *(const uint4*)(base);        // K = khalf*8 + [0,8)
      aF[ms].q[1] = *(const uint4*)(base + 16);   // K = 16 + khalf*8 + [0,8)
    }
#pragma unroll
    for (int ns = 0; ns < 4; ++ns) {
      const bf16* base = sB[buf] + (waveN * 64 + ns * 16 + rsel) * GBK + khalf * 8;
      bF[ns].q[0] = *(const uint4*)(base);        // K = khalf*8 + [0,8)
      bF[ns].q[1] = *(const uint4*)(base + 16);   // K = 16 + khalf*8 + [0,8)
    }
#pragma unroll
    for (int ms = 0; ms < 2; ++ms)
#pragma unroll
      for (int ns = 0; ns < 4; ++ns)
        acc[ms][ns] = __builtin_amdgcn_wmma_f32_16x16x32_bf16(
            false, aF[ms].v, false, bF[ns].v, (short)0, acc[ms][ns],
            false, false);

#if USE_ASYNC_LDS
    __syncthreads();
#else
    if (nk < K) {                    // commit prefetched tile to other buffer
      *(uint4*)(sA[buf ^ 1] + r0 * GBK + kc0)        = rg[0];
      *(uint4*)(sA[buf ^ 1] + (r0 + 64) * GBK + kc0) = rg[1];
      *(uint4*)(sB[buf ^ 1] + r0 * GBK + kc0)        = rg[2];
      *(uint4*)(sB[buf ^ 1] + (r0 + 64) * GBK + kc0) = rg[3];
    }
#endif
    buf ^= 1;
  }

  // ---- epilogue ----
  const int cLane = lane & 15, rHalf = lane >> 4;
#pragma unroll
  for (int ms = 0; ms < 2; ++ms) {
#pragma unroll
    for (int ns = 0; ns < 4; ++ns) {
      long col = colBase + waveN * 64 + ns * 16 + cLane;
      float bv = bias ? bias[col] : 0.f;
#pragma unroll
      for (int v = 0; v < 8; ++v) {
        long row = rowBase + waveM * 32 + ms * 16 + rHalf * 8 + v;
        float val = acc[ms][ns][v] * alpha + bv;
        if (doRelu) val = fmaxf(val, 0.f);
        if (MODE == 0) {
          ((float*)Cp)[(long)blockIdx.z * strideC + row * (long)N + col] = val;
        } else if (MODE == 1) {
          ((bf16*)Cp)[(long)blockIdx.z * strideC + row * (long)N + col] = f2bf(val);
        } else if (MODE == 2) {   // (b*1024+n, h*128+d) -> (b,h,n,d)
          long b = row >> 10, n = row & 1023;
          long h = col >> 7,  d = col & 127;
          ((bf16*)Cp)[(((b * 8 + h) * 1024 + n) << 7) + d] = f2bf(val);
        } else if (MODE == 3) {   // batch z, (n,d) -> (b, n, h*128+d)
          long z = batchOfs + blockIdx.z;
          long b = z >> 3, h = z & 7;
          ((bf16*)Cp)[((b * 1024 + row) << 10) + h * 128 + col] = f2bf(val);
        } else {                  // (b*1024+n, h*128+d) -> (b,h,d,n)  (V transposed)
          long b = row >> 10, n = row & 1023;
          long h = col >> 7,  d = col & 127;
          ((bf16*)Cp)[(((b * 8 + h) * 128 + d) << 10) + n] = f2bf(val);
        }
      }
    }
  }
}

// ---------------- softmax over rows of 1024 f32, emit bf16 probs ----------------
__global__ __launch_bounds__(256) void softmax_rows(const float* __restrict__ S,
                                                    bf16* __restrict__ P) {
  __shared__ float red[256];
  const long row = blockIdx.x;
  const float* s = S + (row << 10);
  bf16* p = P + (row << 10);
  const int tid = threadIdx.x;
  float v[4];
  float m = -3.0e38f;
#pragma unroll
  for (int i = 0; i < 4; ++i) { v[i] = s[tid + i * 256]; m = fmaxf(m, v[i]); }
  red[tid] = m; __syncthreads();
  for (int o = 128; o > 0; o >>= 1) {
    if (tid < o) red[tid] = fmaxf(red[tid], red[tid + o]);
    __syncthreads();
  }
  m = red[0]; __syncthreads();
  float sum = 0.f;
#pragma unroll
  for (int i = 0; i < 4; ++i) { v[i] = __expf(v[i] - m); sum += v[i]; }
  red[tid] = sum; __syncthreads();
  for (int o = 128; o > 0; o >>= 1) {
    if (tid < o) red[tid] += red[tid + o];
    __syncthreads();
  }
  float inv = 1.f / red[0];
#pragma unroll
  for (int i = 0; i < 4; ++i) p[tid + i * 256] = f2bf(v[i] * inv);
}

// ---- per-chunk (one b, all 8 heads) column sum of probabilities over (h, n) ----
__global__ __launch_bounds__(256) void colsum_chunk(const bf16* __restrict__ P,
                                                    float* __restrict__ outCS) {
  int m = blockIdx.x * 256 + threadIdx.x;
  float s = 0.f;
  for (int h = 0; h < 8; ++h) {
    const bf16* base = P + (((long)h) << 20) + m;
    for (int n = 0; n < 1024; ++n) s += bf2f(base[(long)n << 10]);
  }
  outCS[m] = s;
}

// ---------------- residual add + LayerNorm (f32 out, optional bf16 out) ----------------
__global__ __launch_bounds__(256) void add_ln(const float* __restrict__ X,
                                              const float* __restrict__ Y,
                                              const float* __restrict__ g,
                                              const float* __restrict__ bt,
                                              float* __restrict__ outF,
                                              bf16* __restrict__ outB) {
  __shared__ float red[256];
  const long row = blockIdx.x;
  const int tid = threadIdx.x;
  float v[4];
  float s = 0.f;
#pragma unroll
  for (int i = 0; i < 4; ++i) {
    long j = (row << 10) + tid + i * 256;
    v[i] = X[j] + Y[j];
    s += v[i];
  }
  red[tid] = s; __syncthreads();
  for (int o = 128; o > 0; o >>= 1) { if (tid < o) red[tid] += red[tid + o]; __syncthreads(); }
  float mu = red[0] * (1.f / 1024.f);
  __syncthreads();
  float sq = 0.f;
#pragma unroll
  for (int i = 0; i < 4; ++i) { float d = v[i] - mu; sq += d * d; }
  red[tid] = sq; __syncthreads();
  for (int o = 128; o > 0; o >>= 1) { if (tid < o) red[tid] += red[tid + o]; __syncthreads(); }
  float inv = rsqrtf(red[0] * (1.f / 1024.f) + 1e-5f);
#pragma unroll
  for (int i = 0; i < 4; ++i) {
    int c = tid + i * 256;
    float y = (v[i] - mu) * inv * g[c] + bt[c];
    if (outF) outF[(row << 10) + c] = y;
    if (outB) outB[(row << 10) + c] = f2bf(y);
  }
}

// ---------------- halting: a = softmax(colsum/H); halt = hs + a ----------------
__global__ __launch_bounds__(256) void halting_kernel(const float* __restrict__ cs,
                                                      const float* __restrict__ hs,
                                                      float* __restrict__ outHalt,
                                                      float* __restrict__ outA) {
  __shared__ float red[256];
  const int b = blockIdx.x, tid = threadIdx.x;
  float v[4]; float m = -3.0e38f;
#pragma unroll
  for (int i = 0; i < 4; ++i) {
    v[i] = cs[(b << 10) + tid + i * 256] * 0.125f;
    m = fmaxf(m, v[i]);
  }
  red[tid] = m; __syncthreads();
  for (int o = 128; o > 0; o >>= 1) { if (tid < o) red[tid] = fmaxf(red[tid], red[tid + o]); __syncthreads(); }
  m = red[0]; __syncthreads();
  float sum = 0.f;
#pragma unroll
  for (int i = 0; i < 4; ++i) { v[i] = expf(v[i] - m); sum += v[i]; }
  red[tid] = sum; __syncthreads();
  for (int o = 128; o > 0; o >>= 1) { if (tid < o) red[tid] += red[tid + o]; __syncthreads(); }
  float inv = 1.f / red[0];
#pragma unroll
  for (int i = 0; i < 4; ++i) {
    int n = tid + i * 256;
    float a = v[i] * inv;
    outA[(b << 10) + n] = a;
    outHalt[(b << 10) + n] = hs[(b << 10) + n] + a;
  }
}

extern "C" void kernel_launch(void* const* d_in, const int* in_sizes, int n_in,
                              void* d_out, int out_size, void* d_ws, size_t ws_size,
                              hipStream_t stream) {
  const float* x   = (const float*)d_in[0];
  const float* hs  = (const float*)d_in[1];
  const float* Wq  = (const float*)d_in[2];
  const float* Wkv = (const float*)d_in[3];
  const float* Wp  = (const float*)d_in[4];
  const float* bp  = (const float*)d_in[5];
  const float* W1  = (const float*)d_in[6];
  const float* b1  = (const float*)d_in[7];
  const float* W2  = (const float*)d_in[8];
  const float* b2  = (const float*)d_in[9];
  const float* g1  = (const float*)d_in[10];
  const float* be1 = (const float*)d_in[11];
  const float* g2  = (const float*)d_in[12];
  const float* be2 = (const float*)d_in[13];
  (void)in_sizes; (void)n_in; (void)out_size; (void)ws_size;

  char* ws = (char*)d_ws;
  const size_t MiB = 1ull << 20;
  // workspace layout (peak ~649 MiB, lifetime-based reuse)
  bf16*  wqt = (bf16*)(ws + 0 * MiB);      // 2 MiB   Wq^T
  bf16*  wkt = (bf16*)(ws + 2 * MiB);      // 2 MiB   Wk^T
  bf16*  wvt = (bf16*)(ws + 4 * MiB);      // 2 MiB   Wv^T
  bf16*  wpt = (bf16*)(ws + 6 * MiB);      // 2 MiB   Wp^T
  bf16*  w1t = (bf16*)(ws + 8 * MiB);      // 8 MiB   W1^T
  bf16*  w2t = (bf16*)(ws + 16 * MiB);     // 8 MiB   W2^T
  bf16*  xb  = (bf16*)(ws + 24 * MiB);     // 64 MiB (later reused as h_bf16)
  bf16*  Q   = (bf16*)(ws + 88 * MiB);     // 64 MiB  (b,h,n,d)
  bf16*  Kb  = (bf16*)(ws + 152 * MiB);    // 64 MiB  (b,h,n,d)
  bf16*  Vt  = (bf16*)(ws + 216 * MiB);    // 64 MiB  (b,h,d,n)
  float* S   = (float*)(ws + 280 * MiB);   // 32 MiB
  bf16*  P   = (bf16*)(ws + 312 * MiB);    // 16 MiB
  bf16*  OB  = (bf16*)(ws + 328 * MiB);    // 64 MiB  (b,n,c)
  float* O2  = (float*)(ws + 392 * MiB);   // 128 MiB (o@Wp out; later FF2)
  float* H   = (float*)(ws + 520 * MiB);   // 128 MiB
  bf16*  FF1 = (bf16*)(ws + 88 * MiB);     // 256 MiB, reuses dead Q/K/V/S/P/OB
  bf16*  HB  = xb;                         // bf16 h, reuses dead xb
  float* cs  = (float*)(ws + 648 * MiB);   // 128 KiB colsum

  float* outMain = (float*)d_out;
  float* outHalt = outMain + (long)32768 * 1024;
  float* outA    = outHalt + 32768;

  dim3 blk(256);
  const float scale = 0.08838834764831845f;  // 1/sqrt(128)

  // ---- bf16 conversions (+ weight transposes to N x K) ----
  cvt_f32_bf16<<<32768, blk, 0, stream>>>(x, xb, 33554432);
  transpose_cvt<<<dim3(32, 32), blk, 0, stream>>>(Wq, wqt, 1024, 0, 1024);
  transpose_cvt<<<dim3(32, 32), blk, 0, stream>>>(Wkv, wkt, 2048, 0, 1024);
  transpose_cvt<<<dim3(32, 32), blk, 0, stream>>>(Wkv, wvt, 2048, 1024, 1024);
  transpose_cvt<<<dim3(32, 32), blk, 0, stream>>>(Wp, wpt, 1024, 0, 1024);
  transpose_cvt<<<dim3(128, 32), blk, 0, stream>>>(W1, w1t, 4096, 0, 1024);
  transpose_cvt<<<dim3(32, 128), blk, 0, stream>>>(W2, w2t, 1024, 0, 4096);

  // ---- Q/K/V projections ----
  gemm_nt_wmma<2><<<dim3(8, 256, 1), blk, 0, stream>>>(
      xb, 1024, 0, wqt, 1024, 0, Q, 0, nullptr, 1.f, 1024, 1024, 0, 0);
  gemm_nt_wmma<2><<<dim3(8, 256, 1), blk, 0, stream>>>(
      xb, 1024, 0, wkt, 1024, 0, Kb, 0, nullptr, 1.f, 1024, 1024, 0, 0);
  gemm_nt_wmma<4><<<dim3(8, 256, 1), blk, 0, stream>>>(
      xb, 1024, 0, wvt, 1024, 0, Vt, 0, nullptr, 1.f, 1024, 1024, 0, 0);

  // ---- attention: one batch-index b (8 heads) per chunk ----
  for (int c = 0; c < 32; ++c) {
    long bo = (long)c * 8 * 1024 * 128;
    // S = scale * q @ k^T   (K buffer already N x K = keys x d)
    gemm_nt_wmma<0><<<dim3(8, 8, 8), blk, 0, stream>>>(
        Q + bo, 128, 131072, Kb + bo, 128, 131072, S, 1048576, nullptr,
        scale, 1024, 128, 0, 0);
    softmax_rows<<<8192, blk, 0, stream>>>(S, P);
    colsum_chunk<<<4, blk, 0, stream>>>(P, cs + c * 1024);
    // O = P @ V   (Vt is N x K = d x keys)
    gemm_nt_wmma<3><<<dim3(1, 8, 8), blk, 0, stream>>>(
        P, 1024, 1048576, Vt + bo, 1024, 131072, OB, 0, nullptr, 1.f,
        128, 1024, c * 8, 0);
  }

  // ---- output projection + bias ----
  gemm_nt_wmma<0><<<dim3(8, 256, 1), blk, 0, stream>>>(
      OB, 1024, 0, wpt, 1024, 0, O2, 0, bp, 1.f, 1024, 1024, 0, 0);

  // ---- residual + LN1 -> H (f32) and HB (bf16) ----
  add_ln<<<32768, blk, 0, stream>>>(x, O2, g1, be1, H, HB);

  // ---- FFN ----
  gemm_nt_wmma<1><<<dim3(32, 256, 1), blk, 0, stream>>>(
      HB, 1024, 0, w1t, 1024, 0, FF1, 0, b1, 1.f, 4096, 1024, 0, 1);
  gemm_nt_wmma<0><<<dim3(8, 256, 1), blk, 0, stream>>>(
      FF1, 4096, 0, w2t, 4096, 0, O2, 0, b2, 1.f, 1024, 4096, 0, 0);

  // ---- residual + LN2 -> final output ----
  add_ln<<<32768, blk, 0, stream>>>(H, O2, g2, be2, outMain, nullptr);

  // ---- halting score ----
  halting_kernel<<<32, blk, 0, stream>>>(cs, hs, outHalt, outA);
}
